// ZoomAttention_74302934221437
// MI455X (gfx1250) — compile-verified, matrix operand order checked by reading, not run on hardware
//
#include <hip/hip_runtime.h>
#include <hip/hip_bf16.h>

typedef __attribute__((ext_vector_type(16))) _Float16 v16h;
typedef __attribute__((ext_vector_type(8)))  float    v8f;

#define D_MODEL  1024
#define H_HEADS  16
#define DH_HEAD  64
#define NSEQ     2048
#define BATCH    2
#define QK_SCALE 0.03125f  // 1024^-0.5

// ---- CDNA5 async global->LDS copy (ASYNCcnt) --------------------------------
// LDS address = low 32 bits of the generic pointer (flat aperture maps
// addr[31:0] -> LDS offset).
static __device__ __forceinline__ unsigned lds_off(const void* p) {
  return (unsigned)(unsigned long long)p;
}
static __device__ __forceinline__ void async_ld_b128(unsigned ldsoff,
                                                     const void* gptr) {
  asm volatile("global_load_async_to_lds_b128 %0, %1, off"
               :: "v"(ldsoff), "v"(gptr) : "memory");
}
#define S_WAIT_ASYNCCNT(n) asm volatile("s_wait_asynccnt " #n ::: "memory")

static __device__ __forceinline__ v16h load_ab(const _Float16* p) {
  // Per-lane WMMA f16 operand: VGPR0-3 = 8 contiguous f16 at p,
  // VGPR4-7 = 8 contiguous f16 at p+16 (caller bakes in the 8*(lane/16) shift).
  v16h r;
  __builtin_memcpy(&r, p, 16);
  __builtin_memcpy(reinterpret_cast<char*>(&r) + 16, p + 16, 16);
  return r;
}

static __device__ __forceinline__ v8f v8f_zero() {
  v8f z;
#pragma unroll
  for (int j = 0; j < 8; ++j) z[j] = 0.0f;
  return z;
}

static __device__ __forceinline__ v8f wmma_f16(v16h a, v16h b, v8f c) {
  return __builtin_amdgcn_wmma_f32_16x16x32_f16(false, a, false, b, (short)0, c,
                                                false, false);
}

// ---------------- elementwise f32 -> f16 ----------------
__global__ void cvt_f16_kernel(const float* __restrict__ in,
                               _Float16* __restrict__ out, int n) {
  int i = blockIdx.x * blockDim.x + threadIdx.x;
  if (i < n) out[i] = (_Float16)in[i];
}

// transpose+convert: in [K][Nc] f32 row-major -> out [Nc][K] f16 row-major
__global__ void cvtT_f16_kernel(const float* __restrict__ in,
                                _Float16* __restrict__ out, int K, int Nc) {
  int i = blockIdx.x * blockDim.x + threadIdx.x;
  if (i < K * Nc) {
    int n = i / K, k = i - n * K;
    out[i] = (_Float16)in[(size_t)k * Nc + n];
  }
}

// gather V part of qkv into vT[b][h][dh][key] (f16)
__global__ void vtrans_kernel(const _Float16* __restrict__ qkvh,
                              _Float16* __restrict__ vT) {
  int i = blockIdx.x * blockDim.x + threadIdx.x;
  const int total = BATCH * H_HEADS * DH_HEAD * NSEQ;
  if (i < total) {
    int key = i & (NSEQ - 1);
    int r = i >> 11;  // (b*H + h)*DH + dh
    int dh = r & (DH_HEAD - 1); r >>= 6;
    int h = r & (H_HEADS - 1);
    int b = r >> 4;
    vT[i] = qkvh[((size_t)(b * NSEQ) + key) * (3 * D_MODEL) + 2 * D_MODEL +
                 h * DH_HEAD + dh];
  }
}

// ---------------- WMMA GEMM with async-LDS double buffering ------------------
// C[M,Nc] = A[M,K](f16) @ Bt[Nc,K]^T(f16) + bias.
// 128 threads = 4 waves (2x2); workgroup tile 64M x 128N, K-step 32.
template <int OUTF32>
__global__ __launch_bounds__(128) void gemm_wmma_kernel(
    const _Float16* __restrict__ A, const _Float16* __restrict__ Bt,
    const float* __restrict__ bias, void* __restrict__ outp, int M, int Ncols,
    int K) {
  __shared__ _Float16 lsA[2][64][32];
  __shared__ _Float16 lsB[2][128][32];

  const int tid = threadIdx.x;
  const int l = tid & 31, wave = tid >> 5;
  const int lh = l >> 4, lm = l & 15;
  const int wm = wave >> 1, wn = wave & 1;
  const int mbase = blockIdx.y * 64;
  const int nbase = blockIdx.x * 128;

  // stage issue: 2 A-chunks + 4 B-chunks of 16B per thread
  auto issue = [&](int s, int kk) {
#pragma unroll
    for (int i = 0; i < 2; ++i) {
      int c = tid + 128 * i;           // 256 A chunks
      int r = c >> 2, seg = c & 3;
      async_ld_b128(lds_off(&lsA[s][r][seg * 8]),
                    A + (size_t)(mbase + r) * K + kk + seg * 8);
    }
#pragma unroll
    for (int i = 0; i < 4; ++i) {
      int c = tid + 128 * i;           // 512 B chunks
      int r = c >> 2, seg = c & 3;
      async_ld_b128(lds_off(&lsB[s][r][seg * 8]),
                    Bt + (size_t)(nbase + r) * K + kk + seg * 8);
    }
  };

  v8f acc[2][4];
#pragma unroll
  for (int t = 0; t < 2; ++t)
#pragma unroll
    for (int u = 0; u < 4; ++u) acc[t][u] = v8f_zero();

  const int nk = K / 32;
  issue(0, 0);
  for (int ik = 0; ik < nk; ++ik) {
    const int s = ik & 1;
    if (ik + 1 < nk) {
      issue(s ^ 1, (ik + 1) * 32);
      S_WAIT_ASYNCCNT(6);   // stage ik complete (in-order done)
    } else {
      S_WAIT_ASYNCCNT(0);
    }
    __syncthreads();        // all waves' stage-ik data visible in LDS

    v16h a[2], bo[4];
#pragma unroll
    for (int t = 0; t < 2; ++t)
      a[t] = load_ab(&lsA[s][wm * 32 + t * 16 + lm][8 * lh]);
#pragma unroll
    for (int u = 0; u < 4; ++u)
      bo[u] = load_ab(&lsB[s][wn * 64 + u * 16 + lm][8 * lh]);
#pragma unroll
    for (int t = 0; t < 2; ++t)
#pragma unroll
      for (int u = 0; u < 4; ++u) acc[t][u] = wmma_f16(a[t], bo[u], acc[t][u]);

    __syncthreads();        // reads done before buffer s is re-filled
  }

#pragma unroll
  for (int t = 0; t < 2; ++t) {
#pragma unroll
    for (int u = 0; u < 4; ++u) {
      const int col = nbase + wn * 64 + u * 16 + lm;
      const float bv = bias[col];
#pragma unroll
      for (int j = 0; j < 8; ++j) {
        const int row = mbase + wm * 32 + t * 16 + j + 8 * lh;
        const float v = acc[t][u][j] + bv;
        if (OUTF32)
          ((float*)outp)[(size_t)row * Ncols + col] = v;
        else
          ((_Float16*)outp)[(size_t)row * Ncols + col] = (_Float16)v;
      }
    }
  }
}

// ---------------- Flash attention: 4 waves share K/V via async LDS ----------
// Each wave owns 16 query rows; block covers 64 rows of one (b,h).
// S^T = K @ Q^T trick: softmax row = lane%16; S^T C-layout == A K-layout of
// P@V, so P needs no transpose, only f16 packing.
__global__ __launch_bounds__(128) void attn_kernel(
    const _Float16* __restrict__ qkvh, const _Float16* __restrict__ vT,
    const float* __restrict__ gamma, const float* __restrict__ dist,
    _Float16* __restrict__ attnh) {
  __shared__ _Float16 lsK[2][32][64];   // [key_local][dh]
  __shared__ _Float16 lsV[2][64][32];   // [dh][key_local]

  const int tid = threadIdx.x;
  const int l = tid & 31, wave = tid >> 5;
  const int lh = l >> 4, lm = l & 15;
  const int h = blockIdx.y, b = blockIdx.z;
  const int qt = blockIdx.x * 4 + wave;
  const int qrow = qt * 16 + lm;

  const size_t row3d = (size_t)3 * D_MODEL;
  const _Float16* qbase = qkvh + (size_t)(b * NSEQ) * row3d + h * DH_HEAD;
  const _Float16* kbase = qbase + D_MODEL;
  const _Float16* vTbase = vT + (size_t)(b * H_HEADS + h) * DH_HEAD * NSEQ;
  const float g = gamma[b * H_HEADS + h];
  const float* drow = dist + (size_t)qrow * NSEQ;

  // stage issue: 2 K-chunks + 2 V-chunks of 16B per thread
  auto issue = [&](int s, int kb0) {
#pragma unroll
    for (int i = 0; i < 2; ++i) {
      int c = tid + 128 * i;            // 256 K chunks (32 rows x 8)
      int r = c >> 3, seg = c & 7;
      async_ld_b128(lds_off(&lsK[s][r][seg * 8]),
                    kbase + (size_t)(kb0 + r) * row3d + seg * 8);
    }
#pragma unroll
    for (int i = 0; i < 2; ++i) {
      int c = tid + 128 * i;            // 256 V chunks (64 rows x 4)
      int r = c >> 2, seg = c & 3;
      async_ld_b128(lds_off(&lsV[s][r][seg * 8]),
                    vTbase + (size_t)r * NSEQ + kb0 + seg * 8);
    }
  };

  // Q^T as B operand (col = qrow = lane%16, K = dh contiguous): load once.
  v16h bq[2];
#pragma unroll
  for (int c = 0; c < 2; ++c)
    bq[c] = load_ab(qbase + (size_t)qrow * row3d + c * 32 + 8 * lh);

  v8f o[4];
#pragma unroll
  for (int d = 0; d < 4; ++d) o[d] = v8f_zero();
  float m = -1e30f, ssum = 0.0f;

  const int nkb = NSEQ / 32;
  issue(0, 0);
  for (int kb = 0; kb < nkb; ++kb) {
    const int s = kb & 1;
    const int kb0 = kb * 32;
    if (kb + 1 < nkb) {
      issue(s ^ 1, kb0 + 32);
      S_WAIT_ASYNCCNT(4);
    } else {
      S_WAIT_ASYNCCNT(0);
    }
    __syncthreads();

    // S^T tiles: A = 16 key rows from LDS, B = Q^T. Kdim = 64 -> 2 WMMA each.
    v8f st[2];
#pragma unroll
    for (int t = 0; t < 2; ++t) {
      v16h ak0 = load_ab(&lsK[s][t * 16 + lm][8 * lh]);
      v16h ak1 = load_ab(&lsK[s][t * 16 + lm][32 + 8 * lh]);
      st[t] = wmma_f16(ak0, bq[0], v8f_zero());
      st[t] = wmma_f16(ak1, bq[1], st[t]);
    }

    // scale + distance bias; element j of tile t = (key kb0+t*16+8*lh+j, qrow)
    float sc[2][8];
#pragma unroll
    for (int t = 0; t < 2; ++t) {
      float dbuf[8];
      __builtin_memcpy(dbuf, drow + kb0 + t * 16 + 8 * lh, 32);
#pragma unroll
      for (int j = 0; j < 8; ++j)
        sc[t][j] = st[t][j] * QK_SCALE - g * dbuf[j];
    }

    // online softmax (row = qrow = lane%16; cross-half reduce via shfl_xor 16)
    float bm = -1e30f;
#pragma unroll
    for (int t = 0; t < 2; ++t)
#pragma unroll
      for (int j = 0; j < 8; ++j) bm = fmaxf(bm, sc[t][j]);
    bm = fmaxf(bm, __shfl_xor(bm, 16));
    const float nm = fmaxf(m, bm);
    const float alpha = __expf(m - nm);
    float p[2][8];
    float ps = 0.0f;
#pragma unroll
    for (int t = 0; t < 2; ++t)
#pragma unroll
      for (int j = 0; j < 8; ++j) {
        p[t][j] = __expf(sc[t][j] - nm);
        ps += p[t][j];
      }
    ps += __shfl_xor(ps, 16);
    ssum = ssum * alpha + ps;
    m = nm;

    // rescale O accumulators (row of O tile j = qrow-local j+8*lh)
#pragma unroll
    for (int j = 0; j < 8; ++j) {
      const float aj = __shfl(alpha, j + 8 * lh);
#pragma unroll
      for (int d = 0; d < 4; ++d) o[d][j] *= aj;
    }

    // P as A operand: S^T C-layout VGPR j == A K-slot j; just pack to f16.
    v16h pa;
#pragma unroll
    for (int j = 0; j < 8; ++j) {
      pa[j] = (_Float16)p[0][j];
      pa[8 + j] = (_Float16)p[1][j];
    }

    // O += P @ V ; V B-operand from LDS (key contiguous per dh row)
#pragma unroll
    for (int d = 0; d < 4; ++d) {
      v16h vb = load_ab(&lsV[s][d * 16 + lm][8 * lh]);
      o[d] = wmma_f16(pa, vb, o[d]);
    }

    __syncthreads();  // all waves done with buffer s before refill
  }

  const float inv = 1.0f / ssum;
#pragma unroll
  for (int j = 0; j < 8; ++j) {
    const float ij = __shfl(inv, j + 8 * lh);
    const int row = qt * 16 + j + 8 * lh;
#pragma unroll
    for (int d = 0; d < 4; ++d) {
      attnh[((size_t)(b * NSEQ) + row) * D_MODEL + h * DH_HEAD + d * 16 + lm] =
          (_Float16)(o[d][j] * ij);
    }
  }
}

// ---------------------------------------------------------------------------
extern "C" void kernel_launch(void* const* d_in, const int* in_sizes, int n_in,
                              void* d_out, int out_size, void* d_ws,
                              size_t ws_size, hipStream_t stream) {
  const float* x      = (const float*)d_in[0];
  const float* gamma  = (const float*)d_in[1];
  const float* dist   = (const float*)d_in[2];
  const float* qkv_w  = (const float*)d_in[3];
  const float* qkv_b  = (const float*)d_in[4];
  const float* proj_w = (const float*)d_in[5];
  const float* proj_b = (const float*)d_in[6];
  float* out = (float*)d_out;

  const size_t n_x    = (size_t)BATCH * NSEQ * D_MODEL;      // 4Mi elems
  const size_t n_qkvw = (size_t)D_MODEL * 3 * D_MODEL;       // 3Mi
  const size_t n_pw   = (size_t)D_MODEL * D_MODEL;           // 1Mi
  const size_t n_qkv  = (size_t)BATCH * NSEQ * 3 * D_MODEL;  // 12Mi

  char* ws = (char*)d_ws;
  _Float16* xh     = (_Float16*)ws; ws += n_x * 2;
  _Float16* wqkvT  = (_Float16*)ws; ws += n_qkvw * 2;
  _Float16* wprojT = (_Float16*)ws; ws += n_pw * 2;
  _Float16* qkvh   = (_Float16*)ws; ws += n_qkv * 2;
  _Float16* vTh    = (_Float16*)ws; ws += n_x * 2;
  _Float16* attnh  = (_Float16*)ws; ws += n_x * 2;
  // total ws use: ~56 MB

  const int TB = 256;
  cvt_f16_kernel<<<(int)((n_x + TB - 1) / TB), TB, 0, stream>>>(x, xh,
                                                                (int)n_x);
  cvtT_f16_kernel<<<(int)((n_qkvw + TB - 1) / TB), TB, 0, stream>>>(
      qkv_w, wqkvT, D_MODEL, 3 * D_MODEL);
  cvtT_f16_kernel<<<(int)((n_pw + TB - 1) / TB), TB, 0, stream>>>(
      proj_w, wprojT, D_MODEL, D_MODEL);

  dim3 g1(3 * D_MODEL / 128, BATCH * NSEQ / 64);
  gemm_wmma_kernel<0><<<g1, 128, 0, stream>>>(xh, wqkvT, qkv_b, qkvh,
                                              BATCH * NSEQ, 3 * D_MODEL,
                                              D_MODEL);

  vtrans_kernel<<<(int)((n_x + TB - 1) / TB), TB, 0, stream>>>(qkvh, vTh);

  dim3 ga(NSEQ / 64, H_HEADS, BATCH);
  attn_kernel<<<ga, 128, 0, stream>>>(qkvh, vTh, gamma, dist, attnh);

  dim3 g2(D_MODEL / 128, BATCH * NSEQ / 64);
  gemm_wmma_kernel<1><<<g2, 128, 0, stream>>>(attnh, wprojT, proj_b, out,
                                              BATCH * NSEQ, D_MODEL, D_MODEL);
}
